// NCEClassifier_21294447854028
// MI455X (gfx1250) — matrix operand order repaired and checked by reading, not compiled, not run on hardware
//
#include <hip/hip_runtime.h>

#define NB   8192   // num queries
#define NP   8192   // num prototypes
#define NTOT (NB + NP)
#define DIM  1024
#define KSTEP 32
#define NSTEPS (DIM / KSTEP)

typedef __bf16 bf16x8 __attribute__((ext_vector_type(8)));
typedef __bf16 v16bf  __attribute__((ext_vector_type(16)));
typedef float  v8f    __attribute__((ext_vector_type(8)));
typedef int    v4i    __attribute__((ext_vector_type(4)));

// pointer-to-(v4i in AS1/AS3) types for the async-copy builtin
typedef __attribute__((address_space(1))) v4i* as1_v4i;
typedef __attribute__((address_space(3))) v4i* as3_v4i;

union FragU {
    v16bf  v;
    bf16x8 h[2];
};

#if defined(__gfx1250__) && __has_builtin(__builtin_amdgcn_global_load_async_to_lds_b128)
#define USE_ASYNC 1
#else
#define USE_ASYNC 0
#endif

// ---------- bf16 helpers (round-to-nearest-even) ----------
static __device__ __forceinline__ unsigned short f32_to_bf16(float f) {
    unsigned int u = __float_as_uint(f);
    unsigned int lsb = (u >> 16) & 1u;
    u += 0x7fffu + lsb;
    return (unsigned short)(u >> 16);
}
static __device__ __forceinline__ float bf16_to_f32(unsigned short h) {
    return __uint_as_float(((unsigned int)h) << 16);
}

// ---------------------------------------------------------------------------
// Kernel 1: split x (fp32) into bf16 hi/lo planes and compute row sq-norms.
// One 256-thread block per row.
// ---------------------------------------------------------------------------
__global__ void __launch_bounds__(256)
prep_split_norm(const float* __restrict__ x,
                unsigned short* __restrict__ hi,
                unsigned short* __restrict__ lo,
                float* __restrict__ norms) {
    __shared__ float red[256];
    const int row = blockIdx.x;
    const int t   = threadIdx.x;
    const float* xr = x + (size_t)row * DIM;
    unsigned short* hr = hi + (size_t)row * DIM;
    unsigned short* lr = lo + (size_t)row * DIM;

    float s = 0.f;
    #pragma unroll
    for (int c = t; c < DIM; c += 256) {
        float f = xr[c];
        s += f * f;
        unsigned short h = f32_to_bf16(f);
        float fh = bf16_to_f32(h);
        unsigned short l = f32_to_bf16(f - fh);
        hr[c] = h;
        lr[c] = l;
    }
    red[t] = s;
    __syncthreads();
    for (int off = 128; off > 0; off >>= 1) {
        if (t < off) red[t] += red[t + off];
        __syncthreads();
    }
    if (t == 0) norms[row] = red[0];
}

// ---------------------------------------------------------------------------
// Stage one 128-row x 32-col bf16 panel (8 KB) from global to LDS.
// 512 chunks of 16 B; each of the 256 threads moves 2 chunks.
// ---------------------------------------------------------------------------
static __device__ __forceinline__ void stage128(const unsigned short* g,   // at (row0, kb)
                                                unsigned char* lds, int t) {
    #pragma unroll
    for (int rep = 0; rep < 2; ++rep) {
        int idx = t + rep * 256;          // 0..511
        int r = idx >> 2;                 // row 0..127
        int c = idx & 3;                  // 16B chunk 0..3
        const unsigned char* gp = (const unsigned char*)g + (size_t)r * (DIM * 2) + c * 16;
        unsigned char* lp = lds + r * 64 + c * 16;
#if USE_ASYNC
        __builtin_amdgcn_global_load_async_to_lds_b128(
            (as1_v4i)(size_t)gp, (as3_v4i)(size_t)lp, 0, 0);
#else
        *(uint4*)lp = *(const uint4*)gp;
#endif
    }
}

static __device__ __forceinline__ void wait_async0() {
#if USE_ASYNC
#if __has_builtin(__builtin_amdgcn_s_wait_asynccnt)
    __builtin_amdgcn_s_wait_asynccnt(0);
#else
    asm volatile("s_wait_asynccnt 0x0" ::: "memory");
#endif
#endif
}

// ---------------------------------------------------------------------------
// Kernel 2: 128x128 tile GEMM with bf16 hi/lo error-compensated WMMA.
// 8 waves (4 M x 2 N); wave tile 32x64 = 2x4 WMMA tiles.
// acc += hi*hi + hi*lo + lo*hi  (~fp32 accuracy)
// All 12 fragments (4 A + 8 B) are hoisted into registers per k-step so the
// 24 WMMAs issue back-to-back with a single s_wait_dscnt.
// ---------------------------------------------------------------------------
__global__ void __launch_bounds__(256)
nce_gemm(const unsigned short* __restrict__ hi,
         const unsigned short* __restrict__ lo,
         const float* __restrict__ norms,
         const float* __restrict__ temp,
         float* __restrict__ out) {
    // [buf][0]=A_hi [1]=A_lo [2]=B_hi [3]=B_lo, each 128 rows * 64 B
    __shared__ __align__(16) unsigned char smem[2][4][8192];

    const int t    = threadIdx.x;
    const int lane = t & 31;
    const int wave = t >> 5;
    const int wm   = wave & 3;           // M sub-tile of wave: 32*wm
    const int wn   = wave >> 2;          // N sub-tile of wave: 64*wn
    const int bm   = blockIdx.y * 128;
    const int bn   = blockIdx.x * 128;

    const unsigned short* Ahi = hi + (size_t)bm * DIM;
    const unsigned short* Alo = lo + (size_t)bm * DIM;
    const unsigned short* Bhi = hi + (size_t)(NB + bn) * DIM;
    const unsigned short* Blo = lo + (size_t)(NB + bn) * DIM;

    v8f acc[2][4];
    #pragma unroll
    for (int mt = 0; mt < 2; ++mt)
        #pragma unroll
        for (int nt = 0; nt < 4; ++nt)
            acc[mt][nt] = (v8f){0.f, 0.f, 0.f, 0.f, 0.f, 0.f, 0.f, 0.f};

    // prologue: stage k-step 0 into buffer 0
    stage128(Ahi, &smem[0][0][0], t);
    stage128(Alo, &smem[0][1][0], t);
    stage128(Bhi, &smem[0][2][0], t);
    stage128(Blo, &smem[0][3][0], t);
    wait_async0();
    __syncthreads();

    // ISA fragment layouts (16-bit, wave32):
    //  A 16x32: lanes 0-15 row=lane, K 0-7 & 16-23 (bytes 0-15, 32-47)
    //           lanes 16-31 row=lane-16, K 8-15 & 24-31 (bytes 16-31, 48-63)
    //  B 32x16: lane col = lane%16; lanes 0-15 K 0-7 & 16-23 (bytes 0-15, 32-47),
    //           lanes 16-31 K 8-15 & 24-31 (bytes 16-31, 48-63)
    const int lrow = lane & 15;
    const int koA  = (lane < 16) ? 0 : 16;
    const int koB  = (lane < 16) ? 0 : 16;

    for (int s = 0; s < NSTEPS; ++s) {
        const int buf = s & 1;
        if (s + 1 < NSTEPS) {
            const int kb = (s + 1) * KSTEP;
            stage128(Ahi + kb, &smem[buf ^ 1][0][0], t);
            stage128(Alo + kb, &smem[buf ^ 1][1][0], t);
            stage128(Bhi + kb, &smem[buf ^ 1][2][0], t);
            stage128(Blo + kb, &smem[buf ^ 1][3][0], t);
        }

        // ---- load ALL fragments for this k-step into registers ----
        FragU ah[2], al[2], bh[4], bl[4];
        #pragma unroll
        for (int mt = 0; mt < 2; ++mt) {
            const unsigned char* ph = &smem[buf][0][0] + (wm * 32 + mt * 16 + lrow) * 64;
            ah[mt].h[0] = *(const bf16x8*)(ph + koA);
            ah[mt].h[1] = *(const bf16x8*)(ph + 32 + koA);
            const unsigned char* pl = &smem[buf][1][0] + (wm * 32 + mt * 16 + lrow) * 64;
            al[mt].h[0] = *(const bf16x8*)(pl + koA);
            al[mt].h[1] = *(const bf16x8*)(pl + 32 + koA);
        }
        #pragma unroll
        for (int nt = 0; nt < 4; ++nt) {
            const unsigned char* ph = &smem[buf][2][0] + (wn * 64 + nt * 16 + lrow) * 64 + koB;
            bh[nt].h[0] = *(const bf16x8*)(ph);
            bh[nt].h[1] = *(const bf16x8*)(ph + 32);
            const unsigned char* pl = &smem[buf][3][0] + (wn * 64 + nt * 16 + lrow) * 64 + koB;
            bl[nt].h[0] = *(const bf16x8*)(pl);
            bl[nt].h[1] = *(const bf16x8*)(pl + 32);
        }

        // ---- 24 back-to-back WMMAs (distinct accumulators -> no hazards) ----
        #pragma unroll
        for (int nt = 0; nt < 4; ++nt) {
            #pragma unroll
            for (int mt = 0; mt < 2; ++mt) {
                acc[mt][nt] = __builtin_amdgcn_wmma_f32_16x16x32_bf16(
                    false, ah[mt].v, false, bh[nt].v, (short)0, acc[mt][nt], false, false);
                acc[mt][nt] = __builtin_amdgcn_wmma_f32_16x16x32_bf16(
                    false, ah[mt].v, false, bl[nt].v, (short)0, acc[mt][nt], false, false);
                acc[mt][nt] = __builtin_amdgcn_wmma_f32_16x16x32_bf16(
                    false, al[mt].v, false, bh[nt].v, (short)0, acc[mt][nt], false, false);
            }
        }

        if (s + 1 < NSTEPS) wait_async0();
        __syncthreads();
    }

    // Epilogue: scores = (2*qp - |q|^2 - |p|^2) / T
    // C/D layout: VGPR i -> row (i + (lane<16?0:8)), col lane%16
    const float tinv = 1.0f / temp[0];
    const int rowbase = bm + wm * 32;
    const int colbase = bn + wn * 64;
    const int cIn = lane & 15;
    const int rOff = (lane < 16) ? 0 : 8;

    #pragma unroll
    for (int mt = 0; mt < 2; ++mt) {
        #pragma unroll
        for (int nt = 0; nt < 4; ++nt) {
            const int col = colbase + nt * 16 + cIn;
            const float psq = norms[NB + col];
            #pragma unroll
            for (int i = 0; i < 8; ++i) {
                const int row = rowbase + mt * 16 + rOff + i;
                const float qsq = norms[row];
                out[(size_t)row * NP + col] =
                    (2.0f * acc[mt][nt][i] - qsq - psq) * tinv;
            }
        }
    }
}

// ---------------------------------------------------------------------------
extern "C" void kernel_launch(void* const* d_in, const int* in_sizes, int n_in,
                              void* d_out, int out_size, void* d_ws, size_t ws_size,
                              hipStream_t stream) {
    (void)in_sizes; (void)n_in; (void)out_size; (void)ws_size;
    const float* x    = (const float*)d_in[0];
    const float* temp = (const float*)d_in[1];

    unsigned short* hi = (unsigned short*)d_ws;                       // 32 MB
    unsigned short* lo = hi + (size_t)NTOT * DIM;                     // 32 MB
    float* norms       = (float*)(lo + (size_t)NTOT * DIM);           // 64 KB
    float* out         = (float*)d_out;

    prep_split_norm<<<NTOT, 256, 0, stream>>>(x, hi, lo, norms);

    dim3 grid(NP / 128, NB / 128);
    nce_gemm<<<grid, 256, 0, stream>>>(hi, lo, norms, temp, out);
}
